// StructuralSelfAttention_60756607369339
// MI455X (gfx1250) — matrix operand order repaired and checked by reading, not compiled, not run on hardware
//
#include <hip/hip_runtime.h>
#include <hip/hip_bf16.h>

// ---------------------------------------------------------------------------
// Types for CDNA5 WMMA (wave32): v_wmma_f32_16x16x32_bf16
// ---------------------------------------------------------------------------
typedef __attribute__((ext_vector_type(16))) __bf16 v16bf;
typedef __attribute__((ext_vector_type(8)))  float  v8f;

union BFrag { v16bf v; uint4 q[2]; unsigned short u[16]; };
union Rregs { uint4 v[2]; unsigned short s[16]; };

__device__ __forceinline__ unsigned short f32_bf16(float f) {
  unsigned int u = __float_as_uint(f);
  u += 0x7FFFu + ((u >> 16) & 1u);           // round-to-nearest-even
  return (unsigned short)(u >> 16);
}

// ---------------------------------------------------------------------------
// Async global->LDS (CDNA5 GLOBAL_LOAD_ASYNC_TO_LDS_B128, ASYNCcnt-tracked).
// ---------------------------------------------------------------------------
#if defined(__gfx1250__) &&                                        \
    __has_builtin(__builtin_amdgcn_global_load_async_to_lds_b128) && \
    __has_builtin(__builtin_amdgcn_s_wait_asynccnt)
#define USE_ASYNC_LDS 1
#endif

typedef int v4i_b128 __attribute__((vector_size(16)));
typedef __attribute__((address_space(1))) v4i_b128* gas_v4i;
typedef __attribute__((address_space(3))) v4i_b128* las_v4i;

__device__ __forceinline__ void async_copy_b128(const unsigned short* g,
                                                unsigned short* l) {
#if defined(USE_ASYNC_LDS)
  __builtin_amdgcn_global_load_async_to_lds_b128((gas_v4i)g, (las_v4i)l, 0, 0);
#else
  *(uint4*)l = *(const uint4*)g;
#endif
}

// allow the 2 freshly issued b128 copies to stay in flight; older ones done
__device__ __forceinline__ void async_wait_le2() {
#if defined(USE_ASYNC_LDS)
  __builtin_amdgcn_s_wait_asynccnt(2);
#endif
}
__device__ __forceinline__ void async_wait_le0() {
#if defined(USE_ASYNC_LDS)
  __builtin_amdgcn_s_wait_asynccnt(0);
#endif
}

// ---------------------------------------------------------------------------
// f32 -> bf16 conversion
// ---------------------------------------------------------------------------
__global__ __launch_bounds__(256) void cvt_bf16_knl(const float* __restrict__ in,
                                                    unsigned short* __restrict__ out,
                                                    int n) {
  int i = blockIdx.x * 256 + threadIdx.x;
  if (i < n) out[i] = f32_bf16(in[i]);
}

// ---------------------------------------------------------------------------
// Tiled bf16 WMMA GEMM with LDS double buffering + one-tile-ahead prefetch.
// C[M,N] = A[M,K] @ B[K,N] + bias.  128x128 block tile, 8 waves of 64x32.
// A tile: async global->LDS (layout-preserving).  B tile: global->regs early,
// transposed [n][k] store into LDS deferred past the WMMA chain.
// mode bit0: GELU, bit1: store f32, bit2: store bf16
// ---------------------------------------------------------------------------
#define TM 128
#define TN 128
#define TK 32

__global__ __launch_bounds__(256) void gemm_bf16_knl(
    const unsigned short* __restrict__ A, const unsigned short* __restrict__ B,
    const float* __restrict__ bias, float* __restrict__ Cf,
    unsigned short* __restrict__ Cb, int M, int N, int K, int mode) {
  __shared__ __align__(16) unsigned short lA[2][TM * TK];   // [row][k]
  __shared__ __align__(16) unsigned short lBt[2][TN * TK];  // [n][k] transposed

  const int tid  = threadIdx.x;
  const int lane = tid & 31;
  const int wave = tid >> 5;
  const int m0 = blockIdx.y * TM;
  const int n0 = blockIdx.x * TN;
  const int mw = (wave >> 2) * 64;   // 0 or 64
  const int nw = (wave & 3) * 32;    // 0,32,64,96
  const int nl = lane & 15;
  const int g  = lane >> 4;

  v8f acc[4][2];
  for (int mt = 0; mt < 4; ++mt)
    for (int nt = 0; nt < 2; ++nt)
      acc[mt][nt] = (v8f){0.f, 0.f, 0.f, 0.f, 0.f, 0.f, 0.f, 0.f};

  const int arow = tid >> 1, ahalf = tid & 1;   // A tile: 128 rows, 2x16 halves
  const int brow = tid >> 3, bseg  = tid & 7;   // B tile: 32 rows, 8x16 segs
  const unsigned short* gAbase = A + (size_t)(m0 + arow) * K + ahalf * 16;
  unsigned short* const sA0 = &lA[0][arow * TK + ahalf * 16];
  unsigned short* const sA1 = &lA[1][arow * TK + ahalf * 16];

  {  // stage tile 0 into buffer 0
    async_copy_b128(gAbase, sA0);
    async_copy_b128(gAbase + 8, sA0 + 8);
    const unsigned short* gB = B + (size_t)brow * N + n0 + bseg * 16;
    Rregs r;
    r.v[0] = *(const uint4*)gB;
    r.v[1] = *(const uint4*)(gB + 8);
    for (int e = 0; e < 16; ++e)
      lBt[0][(bseg * 16 + e) * TK + brow] = r.s[e];
  }

  const int nsteps = K / TK;
  Rregs rb;
  for (int k = 0; k < nsteps; ++k) {
    const int cur = k & 1, nxt = cur ^ 1;
    const bool more = (k + 1 < nsteps);
    if (more) {  // prefetch tile k+1: async A -> LDS, B -> regs (store later)
      const unsigned short* gA = gAbase + (size_t)(k + 1) * TK;
      unsigned short* sA = nxt ? sA1 : sA0;
      async_copy_b128(gA, sA);
      async_copy_b128(gA + 8, sA + 8);
      const unsigned short* gB =
          B + (size_t)((k + 1) * TK + brow) * N + n0 + bseg * 16;
      rb.v[0] = *(const uint4*)gB;
      rb.v[1] = *(const uint4*)(gB + 8);
      async_wait_le2();   // tile k's async copies have landed (in-order)
    } else {
      async_wait_le0();
    }
    __syncthreads();      // barrier 1: buffer `cur` fully staged everywhere

    BFrag fb[2], fa[4];
    for (int nt = 0; nt < 2; ++nt) {
      const unsigned short* p = &lBt[cur][(nw + nt * 16 + nl) * TK + g * 16];
      fb[nt].q[0] = *(const uint4*)p;
      fb[nt].q[1] = *(const uint4*)(p + 8);
    }
    for (int mt = 0; mt < 4; ++mt) {
      const unsigned short* p = &lA[cur][(mw + mt * 16 + nl) * TK];
      fa[mt].q[0] = *(const uint4*)(p + g * 16);
      fa[mt].q[1] = *(const uint4*)(p + g * 16 + 8);
    }
    for (int mt = 0; mt < 4; ++mt)
      for (int nt = 0; nt < 2; ++nt)
        acc[mt][nt] = __builtin_amdgcn_wmma_f32_16x16x32_bf16(
            false, fa[mt].v, false, fb[nt].v, (short)0, acc[mt][nt], false, false);

    if (more)  // deferred transposed store of tile k+1's B
      for (int e = 0; e < 16; ++e)
        lBt[nxt][(bseg * 16 + e) * TK + brow] = rb.s[e];
    __syncthreads();      // barrier 2: all reads of `cur` done before reuse
  }

  for (int mt = 0; mt < 4; ++mt)
    for (int nt = 0; nt < 2; ++nt)
      for (int i = 0; i < 8; ++i) {
        const int m = m0 + mw + mt * 16 + i + g * 8;
        const int n = n0 + nw + nt * 16 + nl;
        float c = acc[mt][nt][i] + bias[n];
        if (mode & 1) c = 0.5f * c * (1.0f + erff(c * 0.70710678118654752f));
        const size_t idx = (size_t)m * N + n;
        if (mode & 2) Cf[idx] = c;
        if (mode & 4) Cb[idx] = f32_bf16(c);
      }
}

// ---------------------------------------------------------------------------
// Flash attention with structural bias, double-buffered K/V staging.
// One block = (b, h, 128 q rows); 8 waves, each owns 16 q rows.
// K tile: async global->LDS.  V tile: global->regs early, transposed [d][key]
// LDS store deferred past the WMMA work.
// ---------------------------------------------------------------------------
#define AE 1024
#define AD 512
#define AH 8
#define AHD 64

__global__ __launch_bounds__(256) void flash_attn_knl(
    const unsigned short* __restrict__ Qb, const unsigned short* __restrict__ Kb,
    const unsigned short* __restrict__ Vb, const int* __restrict__ rel,
    const unsigned char* __restrict__ kpm, const float* __restrict__ tab,
    unsigned short* __restrict__ Ob) {
  __shared__ __align__(16) unsigned short lK[2][64 * 64];   // [key][d]
  __shared__ __align__(16) unsigned short lVt[2][64 * 64];  // [d][key]
  __shared__ __align__(16) unsigned short lP[8][16 * 64];   // per-wave P tile

  const int tid  = threadIdx.x;
  const int lane = tid & 31;
  const int wave = tid >> 5;
  const int nl = lane & 15;
  const int g  = lane >> 4;
  const int hh = blockIdx.y;
  const int bb = blockIdx.z;
  const int q0 = blockIdx.x * 128 + wave * 16;
  const float scale = 0.125f;  // HD^-0.5

  // Q fragments (16x64 -> two 16x32 A-frags), rows q0..q0+15
  BFrag fq[2];
  for (int f = 0; f < 2; ++f) {
    const unsigned short* qrow =
        Qb + ((size_t)(bb * AE + q0 + nl)) * AD + hh * AHD + f * 32;
    fq[f].q[0] = *(const uint4*)(qrow + g * 16);
    fq[f].q[1] = *(const uint4*)(qrow + g * 16 + 8);
  }

  float m_run[8], l_run[8];
  v8f oacc[4];
  for (int i = 0; i < 8; ++i) { m_run[i] = -3.0e38f; l_run[i] = 0.f; }
  for (int dt = 0; dt < 4; ++dt)
    oacc[dt] = (v8f){0.f, 0.f, 0.f, 0.f, 0.f, 0.f, 0.f, 0.f};

  const int lrow = tid >> 2, lseg = tid & 3;  // cooperative K/V staging
  const unsigned short* gKbase =
      Kb + ((size_t)(bb * AE + lrow)) * AD + hh * AHD + lseg * 16;
  const unsigned short* gVbase =
      Vb + ((size_t)(bb * AE + lrow)) * AD + hh * AHD + lseg * 16;
  unsigned short* const sK0 = &lK[0][lrow * 64 + lseg * 16];
  unsigned short* const sK1 = &lK[1][lrow * 64 + lseg * 16];

  {  // stage kv tile 0 into buffer 0
    async_copy_b128(gKbase, sK0);
    async_copy_b128(gKbase + 8, sK0 + 8);
    Rregs r;
    r.v[0] = *(const uint4*)gVbase;
    r.v[1] = *(const uint4*)(gVbase + 8);
    for (int e = 0; e < 16; ++e)
      lVt[0][(lseg * 16 + e) * 64 + lrow] = r.s[e];
  }

  Rregs rv;
  for (int it = 0; it < AE / 64; ++it) {
    const int cur = it & 1, nxt = cur ^ 1;
    const bool more = (it + 1 < AE / 64);
    const int kv0 = it * 64;
    if (more) {  // prefetch kv tile it+1
      const size_t off = (size_t)(it + 1) * 64 * AD;
      unsigned short* sK = nxt ? sK1 : sK0;
      async_copy_b128(gKbase + off, sK);
      async_copy_b128(gKbase + off + 8, sK + 8);
      rv.v[0] = *(const uint4*)(gVbase + off);
      rv.v[1] = *(const uint4*)(gVbase + off + 8);
      async_wait_le2();
    } else {
      async_wait_le0();
    }
    __syncthreads();      // barrier 1: buffer `cur` staged everywhere

    // S = Q K^T  (4 tiles of 16 keys)
    v8f s[4];
    for (int nt = 0; nt < 4; ++nt) {
      s[nt] = (v8f){0.f, 0.f, 0.f, 0.f, 0.f, 0.f, 0.f, 0.f};
      for (int f = 0; f < 2; ++f) {
        BFrag fk;  // B[k=dim, n=key]: lane nl = key, contiguous dims
        const unsigned short* p = &lK[cur][(nt * 16 + nl) * 64 + f * 32 + g * 16];
        fk.q[0] = *(const uint4*)p;
        fk.q[1] = *(const uint4*)(p + 8);
        s[nt] = __builtin_amdgcn_wmma_f32_16x16x32_bf16(
            false, fq[f].v, false, fk.v, (short)0, s[nt], false, false);
      }
    }

    // scale + structural bias + padding mask (mask hoisted per key column)
    bool mk[4];
    for (int nt = 0; nt < 4; ++nt)
      mk[nt] = kpm[bb * AE + kv0 + nt * 16 + nl] != 0;
    for (int nt = 0; nt < 4; ++nt)
      for (int i = 0; i < 8; ++i) {
        const int qq = q0 + i + g * 8;
        const int kk = kv0 + nt * 16 + nl;
        const int r  = rel[((size_t)bb * AE + qq) * AE + kk];
        float sv = s[nt][i] * scale + tab[r * AH + hh];
        if (mk[nt]) sv = -3.0e38f;
        s[nt][i] = sv;
      }

    // online softmax: per-row reductions across 16-lane halves
    for (int i = 0; i < 8; ++i) {
      float rmax = s[0][i];
      for (int nt = 1; nt < 4; ++nt) rmax = fmaxf(rmax, s[nt][i]);
      for (int off = 8; off; off >>= 1) rmax = fmaxf(rmax, __shfl_xor(rmax, off, 16));
      const float mnew = fmaxf(m_run[i], rmax);
      const float alpha = __expf(m_run[i] - mnew);
      float rsum = 0.f;
      for (int nt = 0; nt < 4; ++nt) {
        const float p = __expf(s[nt][i] - mnew);
        s[nt][i] = p;
        rsum += p;
      }
      for (int off = 8; off; off >>= 1) rsum += __shfl_xor(rsum, off, 16);
      l_run[i] = l_run[i] * alpha + rsum;
      m_run[i] = mnew;
      for (int dt = 0; dt < 4; ++dt) oacc[dt][i] *= alpha;
    }

    // P (accumulator layout) -> LDS -> A-fragment layout.  lP[wave] is
    // wave-private and LDS ops are in-order within a wave: no barrier needed.
    for (int nt = 0; nt < 4; ++nt)
      for (int i = 0; i < 8; ++i)
        lP[wave][(i + g * 8) * 64 + nt * 16 + nl] = f32_bf16(s[nt][i]);

    // O += P V
    for (int f = 0; f < 2; ++f) {
      BFrag fp;
      const unsigned short* pp = lP[wave] + nl * 64 + f * 32;
      fp.q[0] = *(const uint4*)(pp + g * 16);
      fp.q[1] = *(const uint4*)(pp + g * 16 + 8);
      for (int dt = 0; dt < 4; ++dt) {
        BFrag fv;  // B[k=key, n=dim] from transposed V: contiguous keys
        const unsigned short* pv =
            &lVt[cur][(dt * 16 + nl) * 64 + f * 32 + g * 16];
        fv.q[0] = *(const uint4*)pv;
        fv.q[1] = *(const uint4*)(pv + 8);
        oacc[dt] = __builtin_amdgcn_wmma_f32_16x16x32_bf16(
            false, fp.v, false, fv.v, (short)0, oacc[dt], false, false);
      }
    }

    if (more)  // deferred transposed store of tile it+1's V
      for (int e = 0; e < 16; ++e)
        lVt[nxt][(lseg * 16 + e) * 64 + lrow] = rv.s[e];
    __syncthreads();      // barrier 2: all reads of `cur` done before reuse
  }

  // normalize and store in [B, E, H*HD] bf16 layout (feeds O-projection GEMM)
  for (int i = 0; i < 8; ++i) {
    const float inv = (l_run[i] > 0.f) ? 1.0f / l_run[i] : 0.f;
    const int qq = q0 + i + g * 8;
    for (int dt = 0; dt < 4; ++dt) {
      const int dd = dt * 16 + nl;
      Ob[((size_t)(bb * AE + qq)) * AD + hh * AHD + dd] = f32_bf16(oacc[dt][i] * inv);
    }
  }
}

// ---------------------------------------------------------------------------
// Residual add + LayerNorm over D=512; one block per token row.
// ---------------------------------------------------------------------------
__global__ __launch_bounds__(256) void add_ln_knl(
    const float* __restrict__ X, const float* __restrict__ R,
    const float* __restrict__ gm, const float* __restrict__ bt,
    float* __restrict__ outF, unsigned short* __restrict__ outB) {
  const int row = blockIdx.x, tid = threadIdx.x;
  const size_t base = (size_t)row * 512;
  const float v0 = X[base + tid] + R[base + tid];
  const float v1 = X[base + tid + 256] + R[base + tid + 256];
  float s = v0 + v1, sq = v0 * v0 + v1 * v1;
  for (int off = 16; off; off >>= 1) {
    s += __shfl_xor(s, off, 32);
    sq += __shfl_xor(sq, off, 32);
  }
  __shared__ float as_[8], aq_[8];
  if ((tid & 31) == 0) { as_[tid >> 5] = s; aq_[tid >> 5] = sq; }
  __syncthreads();
  s = 0.f; sq = 0.f;
  for (int i = 0; i < 8; ++i) { s += as_[i]; sq += aq_[i]; }
  const float mean = s * (1.0f / 512.0f);
  const float var  = sq * (1.0f / 512.0f) - mean * mean;
  const float rstd = rsqrtf(var + 1e-5f);
  const float o0 = (v0 - mean) * rstd * gm[tid] + bt[tid];
  const float o1 = (v1 - mean) * rstd * gm[tid + 256] + bt[tid + 256];
  outF[base + tid] = o0;
  outF[base + tid + 256] = o1;
  if (outB) {
    outB[base + tid] = f32_bf16(o0);
    outB[base + tid + 256] = f32_bf16(o1);
  }
}

// ---------------------------------------------------------------------------
// Host-side orchestration
// ---------------------------------------------------------------------------
extern "C" void kernel_launch(void* const* d_in, const int* in_sizes, int n_in,
                              void* d_out, int out_size, void* d_ws, size_t ws_size,
                              hipStream_t stream) {
  (void)in_sizes; (void)n_in; (void)out_size; (void)ws_size;
  const float* x   = (const float*)d_in[0];
  const int* rel   = (const int*)d_in[1];
  const unsigned char* kpm = (const unsigned char*)d_in[2];
  const float* Wq = (const float*)d_in[3];  const float* bq = (const float*)d_in[4];
  const float* Wk = (const float*)d_in[5];  const float* bk = (const float*)d_in[6];
  const float* Wv = (const float*)d_in[7];  const float* bv = (const float*)d_in[8];
  const float* Wo = (const float*)d_in[9];  const float* bo = (const float*)d_in[10];
  const float* tab = (const float*)d_in[11];
  const float* g1 = (const float*)d_in[12]; const float* b1n = (const float*)d_in[13];
  const float* g2 = (const float*)d_in[14]; const float* b2n = (const float*)d_in[15];
  const float* W1 = (const float*)d_in[16]; const float* bf1 = (const float*)d_in[17];
  const float* W2 = (const float*)d_in[18]; const float* bf2 = (const float*)d_in[19];

  char* ws = (char*)d_ws;
  // workspace layout (bytes); aliased regions noted
  unsigned short* xb  = (unsigned short*)(ws + 0x0000000);  // 8 MiB (x bf16)
  unsigned short* wqb = (unsigned short*)(ws + 0x0800000);  // 512 KiB each
  unsigned short* wkb = (unsigned short*)(ws + 0x0880000);
  unsigned short* wvb = (unsigned short*)(ws + 0x0900000);
  unsigned short* wob = (unsigned short*)(ws + 0x0980000);
  unsigned short* w1b = (unsigned short*)(ws + 0x0A00000);  // 2 MiB
  unsigned short* w2b = (unsigned short*)(ws + 0x0C00000);  // 2 MiB
  unsigned short* Qb  = (unsigned short*)(ws + 0x0E00000);  // 8 MiB
  unsigned short* Kb  = (unsigned short*)(ws + 0x1600000);  // 8 MiB
  unsigned short* Vb  = (unsigned short*)(ws + 0x1E00000);  // 8 MiB
  float*          x1  = (float*)(ws + 0x2600000);           // 16 MiB
  unsigned short* h1  = (unsigned short*)(ws + 0x3600000);  // 32 MiB
  unsigned short* attnb    = xb;           // reuse: x bf16 dead after QKV
  float*          attnproj = (float*)Qb;   // reuse: Q/K bf16 dead after attention
  unsigned short* x1b      = Vb;           // reuse: V bf16 dead after attention
  float*          ffn2     = attnproj;     // reuse: attnproj dead after LN1

  const dim3 t256(256);
  const int nTok = 8 * 1024;               // 8192 token rows
  const int nx = nTok * 512;

  cvt_bf16_knl<<<(nx + 255) / 256, t256, 0, stream>>>(x, xb, nx);
  cvt_bf16_knl<<<(262144 + 255) / 256, t256, 0, stream>>>(Wq, wqb, 262144);
  cvt_bf16_knl<<<(262144 + 255) / 256, t256, 0, stream>>>(Wk, wkb, 262144);
  cvt_bf16_knl<<<(262144 + 255) / 256, t256, 0, stream>>>(Wv, wvb, 262144);
  cvt_bf16_knl<<<(262144 + 255) / 256, t256, 0, stream>>>(Wo, wob, 262144);
  cvt_bf16_knl<<<(1048576 + 255) / 256, t256, 0, stream>>>(W1, w1b, 1048576);
  cvt_bf16_knl<<<(1048576 + 255) / 256, t256, 0, stream>>>(W2, w2b, 1048576);

  const dim3 g512(512 / TN, nTok / TM);    // (4, 64)
  gemm_bf16_knl<<<g512, t256, 0, stream>>>(xb, wqb, bq, nullptr, Qb, nTok, 512, 512, 4);
  gemm_bf16_knl<<<g512, t256, 0, stream>>>(xb, wkb, bk, nullptr, Kb, nTok, 512, 512, 4);
  gemm_bf16_knl<<<g512, t256, 0, stream>>>(xb, wvb, bv, nullptr, Vb, nTok, 512, 512, 4);

  const dim3 ga(1024 / 128, 8, 8);         // (qblk, head, batch)
  flash_attn_knl<<<ga, t256, 0, stream>>>(Qb, Kb, Vb, rel, kpm, tab, attnb);

  gemm_bf16_knl<<<g512, t256, 0, stream>>>(attnb, wob, bo, attnproj, nullptr,
                                           nTok, 512, 512, 2);
  add_ln_knl<<<nTok, t256, 0, stream>>>(x, attnproj, g1, b1n, x1, x1b);

  const dim3 g2048(2048 / TN, nTok / TM);  // (16, 64)
  gemm_bf16_knl<<<g2048, t256, 0, stream>>>(x1b, w1b, bf1, nullptr, h1,
                                            nTok, 2048, 512, 5);   // GELU + bf16
  gemm_bf16_knl<<<g512, t256, 0, stream>>>(h1, w2b, bf2, ffn2, nullptr,
                                           nTok, 512, 2048, 2);
  add_ln_knl<<<nTok, t256, 0, stream>>>(x1, ffn2, g2, b2n, (float*)d_out, nullptr);
}
